// ResidualSovnet_66477503808121
// MI455X (gfx1250) — compile-verified, matrix operand order checked by reading, not run on hardware
//
#include <hip/hip_runtime.h>
#include <hip/hip_bf16.h>

typedef __bf16 bf16;
typedef __attribute__((ext_vector_type(16))) __bf16 v16bf;
typedef __attribute__((ext_vector_type(8)))  float  v8f;

#define EP_BIAS 1
#define EP_BN   2
#define EP_RELU 4
#define EP_RES  8

static inline int cdiv_i(int a, int b) { return (a + b - 1) / b; }

// ---------------------------------------------------------------------------
// Weight transforms: fold the P4 group action (spatial rot90^r + orientation
// roll) into a materialized bf16 GEMM weight matrix Wt[M=Cout*4][Kpad],
// rows zero-padded to Kpad = ceil(K/32)*32 so conv staging needs no K checks.
// ---------------------------------------------------------------------------
__device__ inline void rot_src(int r, int k, int ky, int kx, int& sy, int& sx) {
    sy = ky; sx = kx;
    for (int t = 0; t < r; ++t) { int ny = sx; int nx = k - 1 - sy; sy = ny; sx = nx; }
}

// stem: w[Cout,Cin,k,k] -> Wt[(co*4+r)][ci*k*k + ky*k + kx]
__global__ void wxform_z2(const float* __restrict__ w, bf16* __restrict__ wt,
                          int Cout, int Cin, int k, int Kpad) {
    int K = Cin * k * k;
    int tot = Cout * 4 * Kpad;
    for (int idx = blockIdx.x * blockDim.x + threadIdx.x; idx < tot;
         idx += gridDim.x * blockDim.x) {
        int kg = idx % Kpad, m = idx / Kpad;
        if (kg >= K) { wt[idx] = (bf16)0.0f; continue; }
        int co = m >> 2, r = m & 3;
        int kx = kg % k, t = kg / k, ky = t % k, ci = t / k;
        int sy, sx; rot_src(r, k, ky, kx, sy, sx);
        wt[idx] = (bf16)w[((co * Cin + ci) * k + sy) * k + sx];
    }
}

// p4: w[Cout,Cin,4,k,k] -> Wt[(co*4+r)][(ci*4+s)*k*k + ky*k + kx]
//     source orientation (s - r) mod 4, spatial rot90^r.
__global__ void wxform_p4(const float* __restrict__ w, bf16* __restrict__ wt,
                          int Cout, int Cin, int k, int Kpad) {
    int k2 = k * k;
    int K = Cin * 4 * k2;
    int tot = Cout * 4 * Kpad;
    for (int idx = blockIdx.x * blockDim.x + threadIdx.x; idx < tot;
         idx += gridDim.x * blockDim.x) {
        int kg = idx % Kpad, m = idx / Kpad;
        if (kg >= K) { wt[idx] = (bf16)0.0f; continue; }
        int co = m >> 2, r = m & 3;
        int rem = kg % k2, ci4 = kg / k2;
        int kx = rem % k, ky = rem / k;
        int ci = ci4 >> 2, s = ci4 & 3;
        int ssrc = (s - r) & 3;
        int sy, sx; rot_src(r, k, ky, kx, sy, sx);
        wt[idx] = (bf16)w[(((co * Cin + ci) * 4 + ssrc) * k + sy) * k + sx];
    }
}

// ---------------------------------------------------------------------------
// Implicit-GEMM P4 conv, one wave32 per 64(M) x 16(N) output tile:
// 4 M-subtiles share one im2col'd B fragment -> 4 WMMAs per staging round.
//   Y[b, m, py, px] = sum_kg Wt[m][kg] * im2col(X)[kg][n]
// A tiles staged via async global->LDS DMA (ASYNCcnt, 8 x b128 per K-step);
// B tile staged via on-the-fly im2col (n-decode carried incrementally in
// SGPRs; kg-decode uses compile-time KK so all div/mod fold to mul/shift).
// bf16 WMMA, fp32 accumulate. Fused epilogue: bias / eval-BN / residual /
// ReLU (params per Cout *channel*: index m>>2, 4 orientations share).
// Requires: M % 64 == 0 (true for every layer here).
// ---------------------------------------------------------------------------
template <int KK>
__global__ __launch_bounds__(32)
void conv_p4_wmma(const bf16* __restrict__ Wt, const float* __restrict__ X,
                  float* __restrict__ Y,
                  int M, int K, int Kpad, int Bn, int Cin4, int H, int W,
                  int Hp, int Wp, int stride, int pad,
                  const float* __restrict__ bias,
                  const float* __restrict__ bng, const float* __restrict__ bnb,
                  const float* __restrict__ bnm, const float* __restrict__ bnv,
                  const float* __restrict__ res, int flags) {
    constexpr int K2 = KK * KK;
    __shared__ __align__(16) bf16 As[64 * 32];   // 4 M-subtiles x (16x32) = 4KB
    __shared__ __align__(16) bf16 Bs[16 * 32];   // 16 cols (N) x 32 (K) = 1KB

    const int lane = threadIdx.x;
    const int m0 = blockIdx.x * 64;
    const int n0 = blockIdx.y * 16;
    const int NS = Bn * Hp * Wp;
    const int CHW = Cin4 * H * W;

    if (lane == 0) __builtin_prefetch(X, 0, 0);          // global_prefetch_b8

    // Block-uniform spatial decode for the 16 N columns: decode n0 once,
    // then carry (px,py,b) incrementally (SALU adds instead of divisions).
    int iy0[16], ix0[16], xb[16], nv[16];
    {
        int px = n0 % Wp; int t = n0 / Wp; int py = t % Hp; int b = t / Hp;
        int xcur = b * CHW;
#pragma unroll
        for (int j = 0; j < 16; ++j) {
            nv[j] = (n0 + j) < NS;
            iy0[j] = py * stride - pad;
            ix0[j] = px * stride - pad;
            xb[j]  = xcur;
            if (++px == Wp) { px = 0; if (++py == Hp) { py = 0; ++b; xcur += CHW; } }
        }
    }

    // LDS byte address of As (flat->LDS: low 32 bits of the generic pointer).
    const unsigned ldsA = (unsigned)(uintptr_t)(&As[0]);

    v8f acc0 = {0.f,0.f,0.f,0.f,0.f,0.f,0.f,0.f};
    v8f acc1 = acc0, acc2 = acc0, acc3 = acc0;

    for (int k0 = 0; k0 < Kpad; k0 += 32) {
        // ---- stage A: async DMA, 8 x b128 per lane-slot, no bounds checks ----
        {
            const bf16* gA = Wt + (size_t)m0 * Kpad + k0;
#pragma unroll
            for (int c = 0; c < 8; ++c) {
                int chunk = lane + 32 * c;                 // 0..255, 16B each
                int ml = chunk >> 2, koff = (chunk & 3) * 8;
                unsigned dst = ldsA + (unsigned)(ml * 64 + koff * 2);
                unsigned long long src =
                    (unsigned long long)(uintptr_t)(gA + (size_t)ml * Kpad + koff);
                asm volatile("global_load_async_to_lds_b128 %0, %1, off"
                             :: "v"(dst), "v"(src) : "memory");
            }
        }
        // ---- stage B: on-the-fly im2col + f32->bf16 (kl == lane) ----
        {
            int kg = k0 + lane;
            int kvalid = (kg < K);
            int kgc = kvalid ? kg : 0;
            int kx = kgc % KK;                 // constant-divisor folds
            int ky = (kgc / KK) % KK;
            int ci = kgc / K2;
#pragma unroll
            for (int j = 0; j < 16; ++j) {
                float v = 0.0f;
                if (nv[j] & kvalid) {
                    int iy = iy0[j] + ky;
                    int ix = ix0[j] + kx;
                    if ((unsigned)iy < (unsigned)H && (unsigned)ix < (unsigned)W)
                        v = X[xb[j] + (ci * H + iy) * W + ix];
                }
                Bs[j * 32 + lane] = (bf16)v;
            }
        }
        asm volatile("s_wait_asynccnt 0x0" ::: "memory");  // A tiles landed
        __syncthreads();

        // ---- fragments per documented 16-bit layout:
        // lane half owns K octet: lanes 0-15 -> K {0..7,16..23}, 16-31 -> +8
        const int row = lane & 15;
        const int kh = (lane >> 4) << 3;
        v16bf bfv;
#pragma unroll
        for (int e = 0; e < 16; ++e) {
            int ke = kh + (e < 8 ? e : e + 8);
            bfv[e] = Bs[row * 32 + ke];
        }
        v16bf af;
#pragma unroll
        for (int e = 0; e < 16; ++e)
            af[e] = As[(0 * 16 + row) * 32 + (kh + (e < 8 ? e : e + 8))];
        acc0 = __builtin_amdgcn_wmma_f32_16x16x32_bf16(false, af, false, bfv,
                                                       (short)0, acc0, false, false);
#pragma unroll
        for (int e = 0; e < 16; ++e)
            af[e] = As[(1 * 16 + row) * 32 + (kh + (e < 8 ? e : e + 8))];
        acc1 = __builtin_amdgcn_wmma_f32_16x16x32_bf16(false, af, false, bfv,
                                                       (short)0, acc1, false, false);
#pragma unroll
        for (int e = 0; e < 16; ++e)
            af[e] = As[(2 * 16 + row) * 32 + (kh + (e < 8 ? e : e + 8))];
        acc2 = __builtin_amdgcn_wmma_f32_16x16x32_bf16(false, af, false, bfv,
                                                       (short)0, acc2, false, false);
#pragma unroll
        for (int e = 0; e < 16; ++e)
            af[e] = As[(3 * 16 + row) * 32 + (kh + (e < 8 ? e : e + 8))];
        acc3 = __builtin_amdgcn_wmma_f32_16x16x32_bf16(false, af, false, bfv,
                                                       (short)0, acc3, false, false);
        __syncthreads();
    }

    // ---- epilogue: C/D layout = VGPR r holds M = r + 8*(lane>=16), N = lane&15
    const int nl = lane & 15;
    const int n = n0 + nl;
    if (n >= NS) return;
    int px = n % Wp; int t = n / Wp; int py = t % Hp; int b = t / Hp;
    const int mh = (lane >> 4) << 3;
    const v8f* accs[4] = {&acc0, &acc1, &acc2, &acc3};
#pragma unroll
    for (int tt = 0; tt < 4; ++tt) {
#pragma unroll
        for (int r = 0; r < 8; ++r) {
            int m = m0 + tt * 16 + mh + r;         // always < M (M % 64 == 0)
            float a = (*accs[tt])[r];
            int c = m >> 2;                        // channel (orientations share)
            if (flags & EP_BIAS) a += bias[c];
            if (flags & EP_BN)
                a = (a - bnm[c]) * rsqrtf(bnv[c] + 1e-5f) * bng[c] + bnb[c];
            size_t oi = ((size_t)(b * M + m) * Hp + py) * Wp + px;
            if (flags & EP_RES) a += res[oi];
            if (flags & EP_RELU) a = fmaxf(a, 0.f);
            Y[oi] = a;
        }
    }
}

// ---------------------------------------------------------------------------
// Primary-capsule LayerNorm over (C,4,H,W) per batch element, in place.
// ---------------------------------------------------------------------------
__global__ void prim_layernorm(float* __restrict__ pc,
                               const float* __restrict__ g,
                               const float* __restrict__ bta, int Nper) {
    __shared__ float rs[256], rq[256];
    int b = blockIdx.x;
    float* p = pc + (size_t)b * Nper;
    float s = 0.f, s2 = 0.f;
    for (int i = threadIdx.x; i < Nper; i += 256) { float v = p[i]; s += v; s2 += v * v; }
    rs[threadIdx.x] = s; rq[threadIdx.x] = s2;
    __syncthreads();
    for (int o = 128; o > 0; o >>= 1) {
        if (threadIdx.x < o) { rs[threadIdx.x] += rs[threadIdx.x + o];
                               rq[threadIdx.x] += rq[threadIdx.x + o]; }
        __syncthreads();
    }
    float mean = rs[0] / Nper;
    float var = rq[0] / Nper - mean * mean;
    float rstd = rsqrtf(var + 1e-5f);
    for (int i = threadIdx.x; i < Nper; i += 256)
        p[i] = (p[i] - mean) * rstd * g[i] + bta[i];
}

// ---------------------------------------------------------------------------
// Capsule "routing": softmax over a singleton axis == ones -> plain sum over
// in_caps, then squash over the 4-orientation axis.
//   preds: [B*ic, oc*16*4, HW]   out: [B, oc, 16, 4, HW]
// ---------------------------------------------------------------------------
__global__ void caps_reduce_squash(const float* __restrict__ preds,
                                   float* __restrict__ out,
                                   int B, int ic, int oc, int HW) {
    int total = B * oc * 16 * HW;
    int idx = blockIdx.x * blockDim.x + threadIdx.x;
    if (idx >= total) return;
    int hw = idx % HW; int t = idx / HW;
    int d = t % 16; t /= 16;
    int o = t % oc; int b = t / oc;
    int M = oc * 64;
    float s[4];
#pragma unroll
    for (int rr = 0; rr < 4; ++rr) {
        int c4 = (o * 16 + d) * 4 + rr;
        float a = 0.f;
        for (int i = 0; i < ic; ++i)
            a += preds[((size_t)(b * ic + i) * M + c4) * HW + hw];
        s[rr] = a;
    }
    float n2 = s[0]*s[0] + s[1]*s[1] + s[2]*s[2] + s[3]*s[3];
    float nn = sqrtf(n2);
    float f = n2 / (1.f + n2) / (nn + 1e-8f);
#pragma unroll
    for (int rr = 0; rr < 4; ++rr)
        out[((((size_t)b * oc + o) * 16 + d) * 4 + rr) * HW + hw] = f * s[rr];
}

// ---------------------------------------------------------------------------
// Head: per (b,class) capsule [16,4] -> transpose/reshape -> 1x1 P4 conv -> max.
// ---------------------------------------------------------------------------
__global__ void head_kernel(const float* __restrict__ caps,
                            const float* __restrict__ lw,
                            const float* __restrict__ lb,
                            float* __restrict__ out, int B) {
    int i = blockIdx.x * blockDim.x + threadIdx.x;
    if (i >= B * 10) return;
    const float* v = caps + (size_t)i * 64;   // [d=16][rr=4]
    float best = -1e30f;
#pragma unroll
    for (int r = 0; r < 4; ++r) {
        float a = lb[0];
        for (int ci = 0; ci < 16; ++ci)
#pragma unroll
            for (int s = 0; s < 4; ++s) {
                int j = ci * 4 + s;                 // flat over transposed [4,16]
                float xv = v[(j & 15) * 4 + (j >> 4)];
                a += xv * lw[ci * 4 + ((s - r) & 3)];
            }
        best = fmaxf(best, a);
    }
    out[i] = best;
}

// ---------------------------------------------------------------------------
// Host orchestration
// ---------------------------------------------------------------------------
extern "C" void kernel_launch(void* const* d_in, const int* in_sizes, int n_in,
                              void* d_out, int out_size, void* d_ws, size_t ws_size,
                              hipStream_t stream) {
    (void)in_sizes; (void)n_in; (void)out_size; (void)ws_size;
    auto F = [&](int i) { return (const float*)d_in[i]; };

    // --- workspace carve (all intermediates are L2-resident on MI455X) ---
    char* p = (char*)d_ws;
    float* A  = (float*)p; p += (size_t)56 << 20;   // big ping (preds up to 52MB)
    float* Bf = (float*)p; p += (size_t)56 << 20;   // big pong
    float* C  = (float*)p; p += (size_t)8  << 20;   // shortcut / small caps
    bf16*  Wb = (bf16*)p;  p += (size_t)20 << 20;   // transformed weights (bf16)

    // --- parameter leaf indices (insertion-order flatten, Nones dropped) ---
    int bw1[7], bbn1[7], bw2[7], bbn2[7], bws = -1, bbns = -1;
    {
        int idx = 6;
        for (int i = 0; i < 7; ++i) {
            bw1[i] = idx++; bbn1[i] = idx; idx += 4;
            bw2[i] = idx++; bbn2[i] = idx; idx += 4;
            if (i == 3) { bws = idx++; bbns = idx; idx += 4; }
        }
    }
    const int I_PRIMW = 81, I_PRIMB = 82, I_LNG = 83, I_LNB = 84;
    const int I_CAPS = 85, I_LINW = 105, I_LINB = 106;

    auto conv = [&](const float* X, float* Y, const float* wraw, bool z2,
                    int Cout, int Cin, int Bn, int Cin4, int H, int W,
                    int k, int stride, int pad,
                    const float* bias, const float* bng, const float* bnb,
                    const float* bnm, const float* bnv,
                    const float* res, int flags) {
        int M = Cout * 4;                       // always a multiple of 64 here
        int K = Cin4 * k * k;
        int Kpad = cdiv_i(K, 32) * 32;
        int tot = M * Kpad;
        if (z2) wxform_z2<<<cdiv_i(tot, 256), 256, 0, stream>>>(wraw, Wb, Cout, Cin, k, Kpad);
        else    wxform_p4<<<cdiv_i(tot, 256), 256, 0, stream>>>(wraw, Wb, Cout, Cin, k, Kpad);
        int Hp = (H + 2 * pad - k) / stride + 1;
        int Wp = (W + 2 * pad - k) / stride + 1;
        int NS = Bn * Hp * Wp;
        dim3 g(M / 64, cdiv_i(NS, 16));
        if (k == 1)
            conv_p4_wmma<1><<<g, 32, 0, stream>>>(Wb, X, Y, M, K, Kpad, Bn, Cin4, H, W,
                                                  Hp, Wp, stride, pad,
                                                  bias, bng, bnb, bnm, bnv, res, flags);
        else
            conv_p4_wmma<3><<<g, 32, 0, stream>>>(Wb, X, Y, M, K, Kpad, Bn, Cin4, H, W,
                                                  Hp, Wp, stride, pad,
                                                  bias, bng, bnb, bnm, bnv, res, flags);
    };

    const int B = 4;
    // --- stem: [4,3,32,32] -> A=[4,256,32,32], BN+ReLU fused ---
    conv(F(0), A, F(1), true, 64, 3, B, 3, 32, 32, 3, 1, 1,
         nullptr, F(2), F(3), F(4), F(5), nullptr, EP_BN | EP_RELU);

    // --- residual blocks (strides 1,1,1,2,1,1,1), in-place on A via Bf ---
    int ch = 64, HH = 32;
    for (int i = 0; i < 7; ++i) {
        int cout = (i >= 3) ? 128 : 64;
        int st = (i == 3) ? 2 : 1;
        int Ho = HH / st;
        if (i == 3)  // projection shortcut: 1x1 stride-2 conv + BN -> C
            conv(A, C, F(bws), false, cout, ch, B, ch * 4, HH, HH, 1, 2, 0,
                 nullptr, F(bbns), F(bbns + 1), F(bbns + 2), F(bbns + 3),
                 nullptr, EP_BN);
        conv(A, Bf, F(bw1[i]), false, cout, ch, B, ch * 4, HH, HH, 3, st, 1,
             nullptr, F(bbn1[i]), F(bbn1[i] + 1), F(bbn1[i] + 2), F(bbn1[i] + 3),
             nullptr, EP_BN | EP_RELU);
        const float* res = (i == 3) ? (const float*)C : (const float*)A;
        conv(Bf, A, F(bw2[i]), false, cout, cout, B, cout * 4, Ho, Ho, 3, 1, 1,
             nullptr, F(bbn2[i]), F(bbn2[i] + 1), F(bbn2[i] + 2), F(bbn2[i] + 3),
             res, EP_BN | EP_RES | EP_RELU);
        ch = cout; HH = Ho;
    }
    // A = [4,512,16,16]

    // --- primary capsules: conv (+bias), then LayerNorm over (C,4,H,W) ---
    conv(A, Bf, F(I_PRIMW), false, 512, 128, B, 512, 16, 16, 3, 1, 1,
         F(I_PRIMB), nullptr, nullptr, nullptr, nullptr, nullptr, EP_BIAS);
    prim_layernorm<<<B, 256, 0, stream>>>(Bf, F(I_LNG), F(I_LNB), 512 * 4 * 16 * 16);
    // Bf = caps [4,32,16,4,16,16] == conv input [128, 64, 16, 16]

    // --- capsule layers: conv (+bias) -> sum over in_caps -> squash(4-axis) ---
    struct { int oc, st, pd; } cfg[5] = {{32,2,0},{32,1,0},{32,1,0},{32,1,1},{10,1,0}};
    float* cin = Bf;
    int Hc = 16;
    for (int L = 0; L < 5; ++L) {
        int oc = cfg[L].oc, st = cfg[L].st, pd = cfg[L].pd;
        int Hp = (Hc + 2 * pd - 3) / st + 1;
        int wi = I_CAPS + 4 * L;
        conv(cin, A, F(wi), false, oc * 16, 16, B * 32, 64, Hc, Hc, 3, st, pd,
             F(wi + 1), nullptr, nullptr, nullptr, nullptr, nullptr, EP_BIAS);
        float* sbuf = (cin == Bf) ? C : Bf;
        int tot = B * oc * 16 * Hp * Hp;
        caps_reduce_squash<<<cdiv_i(tot, 256), 256, 0, stream>>>(A, sbuf, B, 32, oc, Hp * Hp);
        cin = sbuf; Hc = Hp;
    }
    // cin = [4,10,16,4] (HW=1)

    // --- head: 1x1 P4 conv + max over orientations -> [4,10] ---
    head_kernel<<<cdiv_i(B * 10, 64), 64, 0, stream>>>(cin, F(I_LINW), F(I_LINB),
                                                       (float*)d_out, B);
}
